// PCLinear_51144470561334
// MI455X (gfx1250) — matrix operand order repaired and checked by reading, not compiled, not run on hardware
//
#include <hip/hip_runtime.h>
#include <hip/hip_bf16.h>
#include <math.h>

// ---------------------------------------------------------------------------
// CDNA5 (gfx1250) bf16x3 WMMA GEMM pipeline for Polar-Express preconditioned
// linear.  fp32 operands are split hi/lo into pair-packed bf16 planes in LDS
// once per tile; the K-loop runs packed ds_load_b128 fragment loads +
// v_wmma_f32_16x16x32_bf16 (hi*hi + hi*lo + lo*hi).  Global loads for the
// next K-tile are register-prefetched under the current tile's WMMAs.
// ---------------------------------------------------------------------------

typedef __attribute__((ext_vector_type(8)))  float  v8f;
typedef __attribute__((ext_vector_type(16))) __bf16 v16bf;

union Frag {
    v16bf v;
    uint4 q[2];
};

// truncation split: hi = top16(f), lo = top16(f - hi).  Residual ~2^-16 rel.
__device__ __forceinline__ void split2(float f0, float f1,
                                       unsigned& hi, unsigned& lo) {
    unsigned u0 = __float_as_uint(f0), u1 = __float_as_uint(f1);
    unsigned h0 = u0 & 0xFFFF0000u,    h1 = u1 & 0xFFFF0000u;
    float l0 = f0 - __uint_as_float(h0);
    float l1 = f1 - __uint_as_float(h1);
    hi = (u0 >> 16) | h1;
    lo = (__float_as_uint(l0) >> 16) | (__float_as_uint(l1) & 0xFFFF0000u);
}

#define BM 128
#define BN 128
#define BK 32
#define LSTR 20            // LDS row stride in uints (80B: 16B-aligned, bank-spread)

// D[m,n] = alpha * (c0 * (op(A) @ op(B))[m,n] + c1 * S[m,n])
// alpha = *alpha_ptr if non-null else 1.  M%128==0, N%128==0, K%32==0.
template<int TA, int TB>
__global__ __launch_bounds__(256) void gemm_bf16x3_kernel(
    const float* __restrict__ A, const float* __restrict__ B,
    const float* __restrict__ S, float* __restrict__ D,
    int M, int N, int K, int lda, int ldb, int ldc,
    float c0, float c1, const float* __restrict__ alpha_ptr)
{
    // pair-packed bf16 planes: [row][kpair]  (A: row=M, B: row=N)
    __shared__ unsigned Ah[BM * LSTR];
    __shared__ unsigned Al[BM * LSTR];
    __shared__ unsigned Bh[BN * LSTR];
    __shared__ unsigned Bl[BN * LSTR];

    const int tid  = threadIdx.x;
    const int wave = tid >> 5;
    const int lane = tid & 31;
    const int hlf  = lane >> 4;
    const int lrow = lane & 15;
    const int wm   = wave & 3;   // 4 M-waves, 32 rows each
    const int wn   = wave >> 2;  // 2 N-waves, 64 cols each
    const int m0   = blockIdx.y * BM;
    const int n0   = blockIdx.x * BN;

    // staging thread->element maps (compile-time per transpose combo)
    const int sa_r  = TA ? (tid & 127) : (tid >> 1);  // row (M)
    const int sa_g  = TA ? (tid >> 7)  : (tid & 1);   // K group (16 elems)
    const int sb_r  = TB ? (tid >> 1)  : (tid & 127); // row (N)
    const int sb_g  = TB ? (tid & 1)   : (tid >> 7);  // K group

    float fa[16], fb[16];

    auto loadA = [&](int k0) {
        if constexpr (!TA) {
            const float4* p = reinterpret_cast<const float4*>(
                A + (size_t)(m0 + sa_r) * lda + k0 + sa_g * 16);
            #pragma unroll
            for (int j = 0; j < 4; j++) {
                float4 t = p[j];
                fa[4*j] = t.x; fa[4*j+1] = t.y; fa[4*j+2] = t.z; fa[4*j+3] = t.w;
            }
        } else {
            #pragma unroll
            for (int j = 0; j < 16; j++)     // coalesced over m across threads
                fa[j] = A[(size_t)(k0 + sa_g*16 + j) * lda + (m0 + sa_r)];
        }
    };
    auto loadB = [&](int k0) {
        if constexpr (TB) {
            const float4* p = reinterpret_cast<const float4*>(
                B + (size_t)(n0 + sb_r) * ldb + k0 + sb_g * 16);
            #pragma unroll
            for (int j = 0; j < 4; j++) {
                float4 t = p[j];
                fb[4*j] = t.x; fb[4*j+1] = t.y; fb[4*j+2] = t.z; fb[4*j+3] = t.w;
            }
        } else {
            #pragma unroll
            for (int j = 0; j < 16; j++)     // coalesced over n across threads
                fb[j] = B[(size_t)(k0 + sb_g*16 + j) * ldb + (n0 + sb_r)];
        }
    };
    auto storeAB = [&]() {
        #pragma unroll
        for (int j = 0; j < 8; j++)
            split2(fa[2*j], fa[2*j+1], Ah[sa_r*LSTR + sa_g*8 + j],
                                       Al[sa_r*LSTR + sa_g*8 + j]);
        #pragma unroll
        for (int j = 0; j < 8; j++)
            split2(fb[2*j], fb[2*j+1], Bh[sb_r*LSTR + sb_g*8 + j],
                                       Bl[sb_r*LSTR + sb_g*8 + j]);
    };

    v8f acc[2][4] = {};
    const int nk = K / BK;

    loadA(0);
    loadB(0);

    for (int i = 0; i < nk; i++) {
        storeAB();                    // split + ds_store tile i
        __syncthreads();
        if (i + 1 < nk) {             // prefetch tile i+1 under the WMMAs
            loadA((i + 1) * BK);
            loadB((i + 1) * BK);
        }

        // ---- A fragments: 2 M sub-tiles, packed b128 loads ----
        Frag ahi[2], alo[2];
        #pragma unroll
        for (int mt = 0; mt < 2; mt++) {
            int ar = (wm*32 + mt*16 + lrow) * LSTR;
            ahi[mt].q[0] = *reinterpret_cast<const uint4*>(&Ah[ar + hlf*4]);
            ahi[mt].q[1] = *reinterpret_cast<const uint4*>(&Ah[ar + 8 + hlf*4]);
            alo[mt].q[0] = *reinterpret_cast<const uint4*>(&Al[ar + hlf*4]);
            alo[mt].q[1] = *reinterpret_cast<const uint4*>(&Al[ar + 8 + hlf*4]);
        }
        // ---- 4 N sub-tiles per wave ----
        #pragma unroll
        for (int nt = 0; nt < 4; nt++) {
            Frag bhi, blo;
            int br = (wn*64 + nt*16 + lrow) * LSTR;
            bhi.q[0] = *reinterpret_cast<const uint4*>(&Bh[br + hlf*4]);
            bhi.q[1] = *reinterpret_cast<const uint4*>(&Bh[br + 8 + hlf*4]);
            blo.q[0] = *reinterpret_cast<const uint4*>(&Bl[br + hlf*4]);
            blo.q[1] = *reinterpret_cast<const uint4*>(&Bl[br + 8 + hlf*4]);
            #pragma unroll
            for (int mt = 0; mt < 2; mt++) {
                acc[mt][nt] = __builtin_amdgcn_wmma_f32_16x16x32_bf16(
                    false, ahi[mt].v, false, bhi.v, (short)0, acc[mt][nt], false, false);
                acc[mt][nt] = __builtin_amdgcn_wmma_f32_16x16x32_bf16(
                    false, ahi[mt].v, false, blo.v, (short)0, acc[mt][nt], false, false);
                acc[mt][nt] = __builtin_amdgcn_wmma_f32_16x16x32_bf16(
                    false, alo[mt].v, false, bhi.v, (short)0, acc[mt][nt], false, false);
            }
        }
        __syncthreads();              // LDS reuse guard before next store
    }

    const float alpha = alpha_ptr ? *alpha_ptr : 1.0f;
    #pragma unroll
    for (int mt = 0; mt < 2; mt++) {
        #pragma unroll
        for (int nt = 0; nt < 4; nt++) {
            #pragma unroll
            for (int r = 0; r < 8; r++) {
                int m = m0 + wm*32 + mt*16 + hlf*8 + r;  // lanes 16-31 hold M+8
                int n = n0 + wn*64 + nt*16 + lrow;
                float v = c0 * acc[mt][nt][r];
                if (c1 != 0.0f) v += c1 * S[(size_t)m * ldc + n];
                D[(size_t)m * ldc + n] = alpha * v;
            }
        }
    }
}

// ---- deterministic two-pass sum of squares ----
__global__ void sumsq_partial_kernel(const float* __restrict__ X, size_t n,
                                     float* __restrict__ part) {
    __shared__ float sm[256];
    size_t base = (size_t)blockIdx.x * 1024 + threadIdx.x;
    float s = 0.0f;
    #pragma unroll
    for (int j = 0; j < 4; j++) {
        size_t idx = base + (size_t)j * 256;
        if (idx < n) { float v = X[idx]; s += v * v; }
    }
    sm[threadIdx.x] = s;
    __syncthreads();
    for (int off = 128; off > 0; off >>= 1) {
        if (threadIdx.x < (unsigned)off) sm[threadIdx.x] += sm[threadIdx.x + off];
        __syncthreads();
    }
    if (threadIdx.x == 0) part[blockIdx.x] = sm[0];
}

// scal[0]=sumsq, scal[1]=w_norm, scal[2]=1/w_norm, scal[3]=w_norm*gamma
__global__ void finalize_scalars_kernel(const float* __restrict__ part, int nb,
                                        const float* __restrict__ gamma,
                                        float* __restrict__ scal) {
    __shared__ float sm[256];
    float s = 0.0f;
    for (int i = threadIdx.x; i < nb; i += 256) s += part[i];
    sm[threadIdx.x] = s;
    __syncthreads();
    for (int off = 128; off > 0; off >>= 1) {
        if (threadIdx.x < (unsigned)off) sm[threadIdx.x] += sm[threadIdx.x + off];
        __syncthreads();
    }
    if (threadIdx.x == 0) {
        float sumsq = sm[0];
        float fro = sqrtf(sumsq);            // ||gram^2||_F
        float wn = powf(fro, 0.25f) + 1e-8f; // fro^0.25 + eps
        scal[0] = sumsq;
        scal[1] = wn;
        scal[2] = 1.0f / wn;
        scal[3] = wn * gamma[0];             // scale_constant=1
    }
}

__global__ void scale_matrix_kernel(const float* __restrict__ X, float* __restrict__ Y,
                                    size_t n, const float* __restrict__ scal) {
    float inv = scal[2];
    for (size_t i = (size_t)blockIdx.x * blockDim.x + threadIdx.x; i < n;
         i += (size_t)gridDim.x * blockDim.x)
        Y[i] = X[i] * inv;
}

static const float PE_COEF[8][3] = {
    {7.2086f, -15.5131f, 9.0178f}, {3.9623f, -2.5813f, 0.4542f},
    {3.9466f, -2.5765f, 0.4544f}, {3.8991f, -2.5671f, 0.4566f},
    {3.7186f, -2.5308f, 0.4653f}, {3.1390f, -2.3073f, 0.4733f},
    {2.1715f, -1.5246f, 0.3885f}, {1.8648f, -1.2224f, 0.3577f},
};

static inline void launch_gemm(int TA, int TB, dim3 grid, hipStream_t stream,
                               const float* A, const float* B, const float* S,
                               float* D, int M, int N, int K,
                               int lda, int ldb, int ldc,
                               float c0, float c1, const float* alpha_ptr) {
    dim3 blk(256);
    if (!TA && TB)
        gemm_bf16x3_kernel<0,1><<<grid, blk, 0, stream>>>(A,B,S,D,M,N,K,lda,ldb,ldc,c0,c1,alpha_ptr);
    else if (!TA && !TB)
        gemm_bf16x3_kernel<0,0><<<grid, blk, 0, stream>>>(A,B,S,D,M,N,K,lda,ldb,ldc,c0,c1,alpha_ptr);
    else
        gemm_bf16x3_kernel<1,0><<<grid, blk, 0, stream>>>(A,B,S,D,M,N,K,lda,ldb,ldc,c0,c1,alpha_ptr);
}

extern "C" void kernel_launch(void* const* d_in, const int* in_sizes, int n_in,
                              void* d_out, int out_size, void* d_ws, size_t ws_size,
                              hipStream_t stream) {
    const float* x      = (const float*)d_in[0]; // [4,2048,2048]
    const float* weight = (const float*)d_in[1]; // [2048,2048]
    const float* gamma  = (const float*)d_in[2]; // [1]
    float* out = (float*)d_out;

    const int n = 2048;                 // weight is n x n
    const size_t nn = (size_t)n * n;    // 4,194,304
    const int Mlin = in_sizes[0] / n;   // 8192 rows in flattened x

    // d_out (64 MB) doubles as scratch for G and H until the final GEMM.
    float* G = out;            // [n,n]
    float* H = out + nn;       // [n,n]
    // workspace: W ping-pong + scalars + reduction partials (~32 MB + 17 KB)
    char* ws = (char*)d_ws;
    float* W0   = (float*)ws;
    float* W1   = (float*)(ws + nn * sizeof(float));
    float* scal = (float*)(ws + 2 * nn * sizeof(float));
    float* part = scal + 8;    // 4096 partials

    dim3 blk(256);
    dim3 gSq(n / BN, n / BM);  // (16, 16)

    // gram = W @ W^T
    launch_gemm(0, 1, gSq, stream, weight, weight, nullptr, G,
                n, n, n, n, n, n, 1.0f, 0.0f, nullptr);
    // H = gram @ gram
    launch_gemm(0, 0, gSq, stream, G, G, nullptr, H,
                n, n, n, n, n, n, 1.0f, 0.0f, nullptr);
    // w_norm = ||H||_F^0.25 + eps ; also 1/w_norm and w_norm*gamma
    sumsq_partial_kernel<<<(int)(nn / 1024), blk, 0, stream>>>(H, nn, part);
    finalize_scalars_kernel<<<1, blk, 0, stream>>>(part, (int)(nn / 1024), gamma, scal);
    // W0 = W / w_norm
    scale_matrix_kernel<<<4096, blk, 0, stream>>>(weight, W0, nn, scal);

    float* Wc = W0;
    float* Wn = W1;
    for (int i = 0; i < 8; i++) {
        const float a = PE_COEF[i][0], b = PE_COEF[i][1], c = PE_COEF[i][2];
        // G = Wc^T @ Wc
        launch_gemm(1, 0, gSq, stream, Wc, Wc, nullptr, G,
                    n, n, n, n, n, n, 1.0f, 0.0f, nullptr);
        // H = c*(G@G) + b*G     (fused T = b*wtw + c*wtw2)
        launch_gemm(0, 0, gSq, stream, G, G, G, H,
                    n, n, n, n, n, n, c, b, nullptr);
        // Wn = (Wc@H) + a*Wc
        launch_gemm(0, 0, gSq, stream, Wc, H, Wc, Wn,
                    n, n, n, n, n, n, 1.0f, a, nullptr);
        float* t = Wc; Wc = Wn; Wn = t;
    }

    // out = (w_norm * gamma) * X @ Wc^T   (overwrites G/H scratch; reads only x, Wc)
    dim3 gLin(n / BN, Mlin / BM);  // (16, 64)
    launch_gemm(0, 1, gLin, stream, x, Wc, nullptr, out,
                Mlin, n, n, n, n, n, 1.0f, 0.0f, scal + 3);
}
